// MultiHeadAttention_26199300505738
// MI455X (gfx1250) — matrix operand order repaired and checked by reading, not compiled, not run on hardware
//
#include <hip/hip_runtime.h>
#include <hip/hip_bf16.h>

// ---------------------------------------------------------------------------
// MHA block for MI455X (gfx1250): bf16 WMMA GEMMs + flash attention.
// B=4, S=2048, D=2048, NH=16, DQK=DV=128.
// ---------------------------------------------------------------------------

typedef __attribute__((ext_vector_type(16))) __bf16 v16bf;
typedef __attribute__((ext_vector_type(8)))  float  v8f;
typedef __attribute__((ext_vector_type(4)))  int    v4i;

union Frag16 { v16bf v; uint4 q[2]; };

#ifndef __has_builtin
#define __has_builtin(x) 0
#endif
#if __has_builtin(__builtin_amdgcn_global_load_async_to_lds_b128)
#define USE_ASYNC_LDS 1
#else
#define USE_ASYNC_LDS 0
#endif

__device__ __forceinline__ __bf16 to_bf16(float f) {
  unsigned u = __builtin_bit_cast(unsigned, f);
  unsigned r = u + 0x7FFFu + ((u >> 16) & 1u);   // round-to-nearest-even
  unsigned short h = (unsigned short)(r >> 16);
  return __builtin_bit_cast(__bf16, h);
}

__device__ __forceinline__ v8f wmma_bf16(v16bf a, v16bf b, v8f c) {
  return __builtin_amdgcn_wmma_f32_16x16x32_bf16(false, a, false, b,
                                                 (short)0, c, false, false);
}

// 16B global -> LDS copy; async (ASYNCcnt) on gfx1250 toolchains that expose
// the builtin, otherwise synchronous through VGPRs.
__device__ __forceinline__ void cp_lds_b128(void* lds, const void* g) {
#if USE_ASYNC_LDS
  typedef __attribute__((address_space(1))) v4i* gv4p;  // global int4*
  typedef __attribute__((address_space(3))) v4i* lv4p;  // LDS int4*
  // flat->AS1: identical bits; flat->AS3: low 32 bits are the LDS offset
  gv4p gp = reinterpret_cast<gv4p>(reinterpret_cast<unsigned long long>(g));
  lv4p lp = reinterpret_cast<lv4p>(
      (unsigned)reinterpret_cast<unsigned long long>(lds));
  __builtin_amdgcn_global_load_async_to_lds_b128(gp, lp, 0, 0);
#else
  *reinterpret_cast<uint4*>(lds) = *reinterpret_cast<const uint4*>(g);
#endif
}

__device__ __forceinline__ void lds_stage_fence() {
#if USE_ASYNC_LDS
  asm volatile("s_wait_asynccnt 0x0" ::: "memory");
#endif
}

// ---------------------------------------------------------------------------
// Kernel 0a: f32 -> bf16 elementwise convert (activations)
// ---------------------------------------------------------------------------
__global__ void cvt_bf16_kernel(const float* __restrict__ in,
                                __bf16* __restrict__ out, int n) {
  int i = blockIdx.x * blockDim.x + threadIdx.x;
  if (i < n) out[i] = to_bf16(in[i]);
}

// ---------------------------------------------------------------------------
// Kernel 0b: weight transpose + convert: (K x N f32) -> (N x K bf16).
// One-time cost so the GEMM inner loop needs no LDS scatter-transpose.
// ---------------------------------------------------------------------------
__global__ __launch_bounds__(256) void cvt_transpose_kernel(
    const float* __restrict__ in, __bf16* __restrict__ out, int K, int N) {
  __shared__ float t[32][33];
  const int n0 = blockIdx.x * 32, k0 = blockIdx.y * 32;
  const int tx = threadIdx.x & 31, ty = threadIdx.x >> 5;  // ty: 0..7
#pragma unroll
  for (int i = ty; i < 32; i += 8)
    t[i][tx] = in[(size_t)(k0 + i) * N + n0 + tx];
  __syncthreads();
#pragma unroll
  for (int i = ty; i < 32; i += 8)
    out[(size_t)(n0 + i) * K + k0 + tx] = to_bf16(t[tx][i]);
}

// ---------------------------------------------------------------------------
// Kernel 1: C(MxN,f32) = A(MxK,bf16) @ BT(NxK,bf16)^T [+ residual]
// Block tile 128x128, 256 threads (8 waves 4x2), wave tile 32x64 (8 accs).
// Double-buffered LDS, async global->LDS staging.
// ---------------------------------------------------------------------------
#define GT_K 32

__global__ __launch_bounds__(256) void gemm_bf16_wmma(
    const __bf16* __restrict__ A, const __bf16* __restrict__ BT,
    const float* __restrict__ residual, float* __restrict__ C,
    int M, int N, int K) {
  __shared__ __bf16 As[2][128][GT_K];   // 2 x 8 KB
  __shared__ __bf16 Bs[2][128][GT_K];   // 2 x 8 KB

  const int tid   = threadIdx.x;
  const int lane  = tid & 31;
  const int wave  = tid >> 5;
  const int waveM = wave >> 1;          // 0..3
  const int waveN = wave & 1;           // 0..1
  const int m0 = blockIdx.x * 128;
  const int n0 = blockIdx.y * 128;
  const int row16 = lane & 15;
  const int hi    = lane >> 4;

  const int sr = tid >> 1;              // staging row 0..127
  const int sc = (tid & 1) * 16;        // staging col 0 / 16

  v8f acc[2][4] = {};

  auto stage = [&](int bsel, int kk) {
    const __bf16* ap = &A[(size_t)(m0 + sr) * K + kk + sc];
    const __bf16* bp = &BT[(size_t)(n0 + sr) * K + kk + sc];
    cp_lds_b128(&As[bsel][sr][sc],     ap);
    cp_lds_b128(&As[bsel][sr][sc + 8], ap + 8);
    cp_lds_b128(&Bs[bsel][sr][sc],     bp);
    cp_lds_b128(&Bs[bsel][sr][sc + 8], bp + 8);
  };

  int buf = 0;
  stage(buf, 0);
  lds_stage_fence();
  __syncthreads();

  for (int k0 = 0; k0 < K; k0 += GT_K) {
    if (k0 + GT_K < K) {
      stage(buf ^ 1, k0 + GT_K);        // overlap copy of next tile
      if (k0 + 2 * GT_K < K) {          // prefetch the tile after that
        __builtin_prefetch(&A[(size_t)(m0 + sr) * K + k0 + 2 * GT_K + sc], 0, 3);
        __builtin_prefetch(&BT[(size_t)(n0 + sr) * K + k0 + 2 * GT_K + sc], 0, 3);
      }
    }
#pragma unroll
    for (int im = 0; im < 2; ++im) {
      Frag16 a;   // A frag: lane=row, K halves split by lane-half (ISA 7.12.2)
      const int ar = waveM * 32 + im * 16 + row16;
      a.q[0] = *(const uint4*)&As[buf][ar][hi * 8];
      a.q[1] = *(const uint4*)&As[buf][ar][16 + hi * 8];
#pragma unroll
      for (int in = 0; in < 4; ++in) {
        Frag16 b;  // B frag: lane=col, K halves split by lane-half (same as A)
        const int bc = waveN * 64 + in * 16 + row16;
        b.q[0] = *(const uint4*)&Bs[buf][bc][hi * 8];
        b.q[1] = *(const uint4*)&Bs[buf][bc][16 + hi * 8];
        acc[im][in] = wmma_bf16(a.v, b.v, acc[im][in]);
      }
    }
    lds_stage_fence();
    __syncthreads();
    buf ^= 1;
  }

#pragma unroll
  for (int im = 0; im < 2; ++im)
#pragma unroll
    for (int in = 0; in < 4; ++in) {
      const int gm_base = m0 + waveM * 32 + im * 16 + hi * 8;
      const int gn      = n0 + waveN * 64 + in * 16 + row16;
#pragma unroll
      for (int i = 0; i < 8; ++i) {
        const size_t off = (size_t)(gm_base + i) * N + gn;
        float v = acc[im][in][i];
        if (residual) v += residual[off];
        C[off] = v;
      }
    }
}

// ---------------------------------------------------------------------------
// Kernel 2a: RoPE (interleaved pairs) + (B,S,H,Dh)->(B,H,S,Dh), f32->bf16
// ---------------------------------------------------------------------------
__global__ void rope_transpose_kernel(const float* __restrict__ X,
                                      const float* __restrict__ cosT,
                                      const float* __restrict__ sinT,
                                      __bf16* __restrict__ Y) {
  const int S = 2048, NH = 16;
  int idx = blockIdx.x * blockDim.x + threadIdx.x;   // ((b*S+s)*NH+h)*64+p
  int p = idx & 63;
  int h = (idx >> 6) & 15;
  int s = (idx >> 10) & 2047;
  int b = idx >> 21;
  const float* src = X + ((size_t)b * S + s) * 2048 + h * 128 + 2 * p;
  float x1 = src[0], x2 = src[1];
  float c = cosT[s * 64 + p], sn = sinT[s * 64 + p];
  __bf16* dst = Y + (((size_t)b * NH + h) * S + s) * 128 + 2 * p;
  dst[0] = to_bf16(x1 * c - x2 * sn);
  dst[1] = to_bf16(x1 * sn + x2 * c);
}

// ---------------------------------------------------------------------------
// Kernel 2b: V: (B,S,H*Dh f32) -> transposed (B,H,Dh,S bf16), LDS-tiled
// ---------------------------------------------------------------------------
__global__ __launch_bounds__(256) void v_transpose_kernel(
    const float* __restrict__ X, __bf16* __restrict__ YT) {
  const int S = 2048, DH = 128;
  __shared__ float t[32][33];
  const int s0 = blockIdx.x * 32, d0 = blockIdx.y * 32;
  const int b = blockIdx.z >> 4, h = blockIdx.z & 15;
  const int tx = threadIdx.x & 31, ty = threadIdx.x >> 5;
  const float* src = X + (size_t)b * S * 2048 + h * DH;
#pragma unroll
  for (int i = ty; i < 32; i += 8)
    t[i][tx] = src[(size_t)(s0 + i) * 2048 + d0 + tx];
  __syncthreads();
  __bf16* dst = YT + ((size_t)(b * 16 + h) * DH) * S;
#pragma unroll
  for (int i = ty; i < 32; i += 8)
    dst[(size_t)(d0 + i) * S + s0 + tx] = to_bf16(t[tx][i]);
}

// ---------------------------------------------------------------------------
// Kernel 3: flash attention. 128 threads = 4 waves; wave owns a 16-row Q
// tile; 32-key blocks staged in LDS (K natural from Kr, V from pre-transposed
// VrT) with async copies.
// ---------------------------------------------------------------------------
__global__ __launch_bounds__(128) void flash_attn_kernel(
    const __bf16* __restrict__ Q, const __bf16* __restrict__ K,
    const __bf16* __restrict__ VT, const unsigned char* __restrict__ mask,
    __bf16* __restrict__ O) {
  const int S = 2048, DH = 128, NH = 16;
  __shared__ __bf16 Ks[32][DH];     // 8 KB   (rows = key, cols = d)
  __shared__ __bf16 VsT[DH][32];    // 8 KB   (rows = d, cols = key)
  __shared__ __bf16 Ps[4][16][32];  // 4 KB   per-wave P transpose scratch

  const int tid  = threadIdx.x;
  const int lane = tid & 31;
  const int wave = tid >> 5;
  const int row16 = lane & 15;
  const int hi    = lane >> 4;
  const int h = blockIdx.y, b = blockIdx.z;
  const size_t base   = ((size_t)b * NH + h) * (size_t)S * DH;  // Kr rows
  const size_t baseT  = ((size_t)b * NH + h) * (size_t)DH * S;  // VrT rows
  const int q0 = blockIdx.x * 64 + wave * 16;
  const unsigned char* mk = mask + (size_t)b * S;   // (B,1,1,S) bool
  const float scale = 0.08838834764831845f;         // 1/sqrt(128)

  // Q A-fragments: resident for the whole key loop (4 x 16x32 bf16)
  Frag16 qf[4];
  {
    const __bf16* qrow = Q + base + (size_t)(q0 + row16) * DH;
#pragma unroll
    for (int f = 0; f < 4; ++f) {
      qf[f].q[0] = *(const uint4*)&qrow[f * 32 + hi * 8];
      qf[f].q[1] = *(const uint4*)&qrow[f * 32 + 16 + hi * 8];
    }
  }

  float mrow[8], lrow[8];
#pragma unroll
  for (int i = 0; i < 8; ++i) { mrow[i] = -3.0e38f; lrow[i] = 0.f; }
  v8f Oacc[8] = {};

  for (int kb = 0; kb < S / 32; ++kb) {
    const int key0 = kb * 32;
    // stage K (32x128) and V^T (128x32): 512 + 512 x 16B chunks, async
#pragma unroll
    for (int it = 0; it < 4; ++it) {
      const int c = tid + it * 128;
      const int kr = c >> 4;                 // key row 0..31
      const int kd = (c & 15) * 8;           // d offset
      cp_lds_b128(&Ks[kr][kd], &K[base + (size_t)(key0 + kr) * DH + kd]);
      const int vd = c >> 2;                 // d row 0..127
      const int vk = (c & 3) * 8;            // key offset
      cp_lds_b128(&VsT[vd][vk], &VT[baseT + (size_t)vd * S + key0 + vk]);
    }
    lds_stage_fence();
    __syncthreads();

    // scores: two 16x16 key tiles, K-dim 128 via 4 WMMAs each
    v8f sc0 = {}, sc1 = {};
#pragma unroll
    for (int f = 0; f < 4; ++f) {
      Frag16 b0, b1;
      b0.q[0] = *(const uint4*)&Ks[row16][f * 32 + hi * 8];
      b0.q[1] = *(const uint4*)&Ks[row16][f * 32 + 16 + hi * 8];
      b1.q[0] = *(const uint4*)&Ks[16 + row16][f * 32 + hi * 8];
      b1.q[1] = *(const uint4*)&Ks[16 + row16][f * 32 + 16 + hi * 8];
      sc0 = wmma_bf16(qf[f].v, b0.v, sc0);
      sc1 = wmma_bf16(qf[f].v, b1.v, sc1);
    }

    // online softmax (lane owns one key column, 8 q-rows)
    const bool mA = mk[key0 + row16] != 0;
    const bool mB = mk[key0 + 16 + row16] != 0;
    float pr0[8], pr1[8], alpha[8];
#pragma unroll
    for (int i = 0; i < 8; ++i) {
      float s0 = mA ? -3.0e38f : sc0[i] * scale;
      float s1 = mB ? -3.0e38f : sc1[i] * scale;
      float mb = fmaxf(s0, s1);
#pragma unroll
      for (int off = 8; off; off >>= 1) mb = fmaxf(mb, __shfl_xor(mb, off, 32));
      float mnew = fmaxf(mrow[i], mb);
      float a  = (mnew <= -1.0e38f) ? 1.f : __expf(mrow[i] - mnew);
      float p0 = (s0 <= -1.0e38f)   ? 0.f : __expf(s0 - mnew);
      float p1 = (s1 <= -1.0e38f)   ? 0.f : __expf(s1 - mnew);
      float ps = p0 + p1;
#pragma unroll
      for (int off = 8; off; off >>= 1) ps += __shfl_xor(ps, off, 32);
      lrow[i] = lrow[i] * a + ps;
      mrow[i] = mnew;
      alpha[i] = a;
      pr0[i] = p0;
      pr1[i] = p1;
    }
#pragma unroll
    for (int c = 0; c < 8; ++c)
#pragma unroll
      for (int i = 0; i < 8; ++i) Oacc[c][i] *= alpha[i];

    // transpose P (col-owned -> row-owned A fragment) via per-wave LDS
#pragma unroll
    for (int i = 0; i < 8; ++i) {
      Ps[wave][i + 8 * hi][row16]      = to_bf16(pr0[i]);
      Ps[wave][i + 8 * hi][16 + row16] = to_bf16(pr1[i]);
    }
    Frag16 pf;   // same-wave LDS RAW: DS ops are in-order per wave
    pf.q[0] = *(const uint4*)&Ps[wave][row16][hi * 8];
    pf.q[1] = *(const uint4*)&Ps[wave][row16][16 + hi * 8];

    // O += P(16x32) @ V(32x128)
#pragma unroll
    for (int c = 0; c < 8; ++c) {
      Frag16 bv;
      bv.q[0] = *(const uint4*)&VsT[c * 16 + row16][hi * 8];
      bv.q[1] = *(const uint4*)&VsT[c * 16 + row16][16 + hi * 8];
      Oacc[c] = wmma_bf16(pf.v, bv.v, Oacc[c]);
    }
    __syncthreads();
  }

  // normalize and store O (B,S,NH*DV) bf16 for the out-projection GEMM
  float rl[8];
#pragma unroll
  for (int i = 0; i < 8; ++i) rl[i] = (lrow[i] > 0.f) ? 1.f / lrow[i] : 0.f;
#pragma unroll
  for (int c = 0; c < 8; ++c)
#pragma unroll
    for (int i = 0; i < 8; ++i) {
      const int srow = q0 + i + 8 * hi;
      const int col  = h * DH + c * 16 + row16;
      O[((size_t)b * S + srow) * 2048 + col] = to_bf16(Oacc[c][i] * rl[i]);
    }
}

// ---------------------------------------------------------------------------
// Kernel 4: LayerNorm over D=2048, one 256-thread block per row
// ---------------------------------------------------------------------------
__global__ __launch_bounds__(256) void layernorm_kernel(
    const float* __restrict__ X, const float* __restrict__ gamma,
    const float* __restrict__ beta, float* __restrict__ Y, int D) {
  __shared__ float red[2][8];
  const int row = blockIdx.x;
  const float* x = X + (size_t)row * D;
  float s = 0.f, s2 = 0.f;
  for (int i = threadIdx.x; i < D; i += 256) {
    float v = x[i];
    s += v;
    s2 += v * v;
  }
#pragma unroll
  for (int off = 16; off; off >>= 1) {
    s  += __shfl_xor(s, off, 32);
    s2 += __shfl_xor(s2, off, 32);
  }
  const int lane = threadIdx.x & 31, wv = threadIdx.x >> 5;
  if (lane == 0) { red[0][wv] = s; red[1][wv] = s2; }
  __syncthreads();
  if (wv == 0) {
    s  = (lane < 8) ? red[0][lane] : 0.f;
    s2 = (lane < 8) ? red[1][lane] : 0.f;
#pragma unroll
    for (int off = 4; off; off >>= 1) {
      s  += __shfl_xor(s, off, 32);
      s2 += __shfl_xor(s2, off, 32);
    }
    if (lane == 0) { red[0][0] = s; red[1][0] = s2; }
  }
  __syncthreads();
  const float mean = red[0][0] / (float)D;
  const float var  = red[1][0] / (float)D - mean * mean;
  const float inv  = rsqrtf(var + 1e-5f);
  for (int i = threadIdx.x; i < D; i += 256)
    Y[(size_t)row * D + i] = (x[i] - mean) * inv * gamma[i] + beta[i];
}

// ---------------------------------------------------------------------------
// Host orchestration
// ---------------------------------------------------------------------------
extern "C" void kernel_launch(void* const* d_in, const int* in_sizes, int n_in,
                              void* d_out, int out_size, void* d_ws,
                              size_t ws_size, hipStream_t stream) {
  (void)in_sizes; (void)n_in; (void)out_size; (void)ws_size;
  const int Bb = 4, Ss = 2048, Dd = 2048, NHn = 16;
  const size_t MS = (size_t)Bb * Ss;            // 8192 rows

  const float* q_in  = (const float*)d_in[0];
  const float* kv_in = (const float*)d_in[1];
  const unsigned char* mask = (const unsigned char*)d_in[2];  // jnp bool_
  const float* Wq = (const float*)d_in[3];
  const float* Wk = (const float*)d_in[4];
  const float* Wv = (const float*)d_in[5];
  const float* Wo = (const float*)d_in[6];
  const float* gamma = (const float*)d_in[7];
  const float* beta  = (const float*)d_in[8];
  const float* cosT  = (const float*)d_in[9];
  const float* sinT  = (const float*)d_in[10];
  float* out = (float*)d_out;

  char* ws = (char*)d_ws;
  size_t off = 0;
  auto carve = [&](size_t bytes) -> void* {
    void* p = ws + off;
    off += (bytes + 255) & ~(size_t)255;
    return p;
  };
  __bf16* qa   = (__bf16*)carve(MS * Dd * 2);
  __bf16* kva  = (__bf16*)carve(MS * Dd * 2);
  __bf16* wqbT = (__bf16*)carve((size_t)Dd * Dd * 2);
  __bf16* wkbT = (__bf16*)carve((size_t)Dd * Dd * 2);
  __bf16* wvbT = (__bf16*)carve((size_t)Dd * Dd * 2);
  __bf16* wobT = (__bf16*)carve((size_t)Dd * Dd * 2);
  float*  raw  = (float*)carve(MS * Dd * 4);   // reused: Q/K/V raw, then proj
  __bf16* Qr   = (__bf16*)carve(MS * Dd * 2);
  __bf16* Kr   = (__bf16*)carve(MS * Dd * 2);
  __bf16* VrT  = (__bf16*)carve(MS * Dd * 2);
  __bf16* Ob   = (__bf16*)carve(MS * Dd * 2);

  // 0) precision conversion (+ one-time weight transposes)
  const int actN = (int)(MS * Dd);
  cvt_bf16_kernel<<<(actN + 255) / 256, 256, 0, stream>>>(q_in, qa, actN);
  cvt_bf16_kernel<<<(actN + 255) / 256, 256, 0, stream>>>(kv_in, kva, actN);
  dim3 gT(Dd / 32, Dd / 32);
  cvt_transpose_kernel<<<gT, 256, 0, stream>>>(Wq, wqbT, Dd, Dd);
  cvt_transpose_kernel<<<gT, 256, 0, stream>>>(Wk, wkbT, Dd, Dd);
  cvt_transpose_kernel<<<gT, 256, 0, stream>>>(Wv, wvbT, Dd, Dd);
  cvt_transpose_kernel<<<gT, 256, 0, stream>>>(Wo, wobT, Dd, Dd);

  dim3 gGemm((unsigned)(MS / 128), (unsigned)(Dd / 128));
  const int ropeBlocks = (int)(MS * NHn * 64 / 256);

  // 1) Q projection + RoPE
  gemm_bf16_wmma<<<gGemm, 256, 0, stream>>>(qa, wqbT, nullptr, raw,
                                            (int)MS, Dd, Dd);
  rope_transpose_kernel<<<ropeBlocks, 256, 0, stream>>>(raw, cosT, sinT, Qr);
  // 2) K projection + RoPE
  gemm_bf16_wmma<<<gGemm, 256, 0, stream>>>(kva, wkbT, nullptr, raw,
                                            (int)MS, Dd, Dd);
  rope_transpose_kernel<<<ropeBlocks, 256, 0, stream>>>(raw, cosT, sinT, Kr);
  // 3) V projection + transpose to (B,H,Dh,S)
  gemm_bf16_wmma<<<gGemm, 256, 0, stream>>>(kva, wvbT, nullptr, raw,
                                            (int)MS, Dd, Dd);
  v_transpose_kernel<<<dim3(Ss / 32, 128 / 32, Bb * NHn), 256, 0, stream>>>(
      raw, VrT);

  // 4) attention
  flash_attn_kernel<<<dim3(Ss / 64, NHn, Bb), 128, 0, stream>>>(Qr, Kr, VrT,
                                                                mask, Ob);
  // 5) out projection + residual (f32 epilogue)
  gemm_bf16_wmma<<<gGemm, 256, 0, stream>>>(Ob, wobT, q_in, raw,
                                            (int)MS, Dd, Dd);
  // 6) layernorm
  layernorm_kernel<<<(unsigned)MS, 256, 0, stream>>>(raw, gamma, beta, out, Dd);
}
